// GPS_PascalVOC_34832184770969
// MI455X (gfx1250) — matrix operand order repaired and checked by reading, not compiled, not run on hardware
//
#include <hip/hip_runtime.h>
#include <hip/hip_bf16.h>
#include <math.h>

// ---------------------------------------------------------------------------
// Types for WMMA fragments (gfx1250, wave32)
// ---------------------------------------------------------------------------
typedef __attribute__((ext_vector_type(16))) __bf16          v16bf;
typedef __attribute__((ext_vector_type(8)))  float           v8f;
typedef __attribute__((ext_vector_type(16))) unsigned short  v16u16;
typedef __attribute__((ext_vector_type(8)))  unsigned short  v8u16;

union Frag {
    v16u16 u;
    v16bf  bf;
    unsigned short s[16];
};

__device__ __forceinline__ unsigned short f2bf(float f) {
    union { float f; unsigned int u; } v; v.f = f;
    unsigned int r = v.u + 0x7FFFu + ((v.u >> 16) & 1u);   // round-to-nearest-even
    return (unsigned short)(r >> 16);
}

__device__ __forceinline__ v8f wmma_bf16(const Frag& a, const Frag& b, v8f c) {
    return __builtin_amdgcn_wmma_f32_16x16x32_bf16(
        /*neg_a=*/false, a.bf, /*neg_b=*/false, b.bf,
        /*c_mod=*/(short)0, c, /*reuse_a=*/false, /*reuse_b=*/false);
}

// ---------------------------------------------------------------------------
// gfx1250 async global->LDS copy (ASYNCcnt-tracked), 16B per lane.
// lds_dst low 32 bits of the generic pointer are the LDS byte offset.
// ---------------------------------------------------------------------------
__device__ __forceinline__ void async_copy_b128(const unsigned short* lds_dst,
                                                const unsigned short* gsrc) {
    unsigned int loff = (unsigned int)(size_t)lds_dst;
    asm volatile("global_load_async_to_lds_b128 %0, %1, off"
                 :: "v"(loff), "v"(gsrc)
                 : "memory");
}

__device__ __forceinline__ void wait_async0() {
#if __has_builtin(__builtin_amdgcn_s_wait_asynccnt)
    __builtin_amdgcn_s_wait_asynccnt(0);
#else
    asm volatile("s_wait_asynccnt 0x0" ::: "memory");
#endif
}

// ---------------------------------------------------------------------------
// Generic GEMM:  C[M,N] = act( A[M,K] @ B + bias )
//   B is [K,N] row-major if bT==0, or [N,K] row-major (i.e. A @ B^T) if bT==1.
// Block: 128 threads (4 waves). Tile: 64 rows x 64 cols. K step: 32 (bf16).
// ---------------------------------------------------------------------------
__global__ void k_gemm(const float* __restrict__ A, const float* __restrict__ B,
                       const float* __restrict__ bias, float* __restrict__ C,
                       int M, int N, int K, int bT, int relu)
{
    __shared__ __align__(32) unsigned short ldsA[64 * 32];   // [row][k]
    __shared__ __align__(32) unsigned short ldsBt[64 * 32];  // [col][k]

    const int row0 = blockIdx.x * 64;
    const int nb   = blockIdx.y * 64;
    const int tid  = threadIdx.x;
    const int wave = tid >> 5;
    const int lane = tid & 31;

    v8f acc[4] = {};   // 4 N-tiles of 16x16

    for (int kk = 0; kk < K; kk += 32) {
        __syncthreads();
        for (int i = tid; i < 64 * 32; i += 128) {
            int r = i >> 5, c = i & 31;
            ldsA[r * 32 + c] = f2bf(A[(size_t)(row0 + r) * K + kk + c]);
        }
        if (!bT) {
            for (int i = tid; i < 32 * 64; i += 128) {
                int k = i >> 6, n = i & 63;
                ldsBt[n * 32 + k] = f2bf(B[(size_t)(kk + k) * N + nb + n]);
            }
        } else {
            for (int i = tid; i < 64 * 32; i += 128) {
                int n = i >> 5, k = i & 31;
                ldsBt[n * 32 + k] = f2bf(B[(size_t)(nb + n) * K + kk + k]);
            }
        }
        __syncthreads();

        const int arow = wave * 16 + (lane & 15);
        const int ak0  = (lane >> 4) * 16;
        Frag a;
        const unsigned short* ap = &ldsA[arow * 32 + ak0];
        *(v8u16*)&a.s[0] = *(const v8u16*)ap;
        *(v8u16*)&a.s[8] = *(const v8u16*)(ap + 8);

        #pragma unroll
        for (int nt = 0; nt < 4; ++nt) {
            Frag b;
            const int col = nt * 16 + (lane & 15);
            const int bk0 = (lane >> 4) * 16;
            b.u = *(const v16u16*)&ldsBt[col * 32 + bk0];
            acc[nt] = wmma_bf16(a, b, acc[nt]);
        }
    }

    #pragma unroll
    for (int nt = 0; nt < 4; ++nt) {
        #pragma unroll
        for (int r = 0; r < 8; ++r) {
            int row = row0 + wave * 16 + r + ((lane >> 4) << 3);
            int col = nb + nt * 16 + (lane & 15);
            float v = acc[nt][r] + (bias ? bias[col] : 0.f);
            if (relu) v = fmaxf(v, 0.f);
            C[(size_t)row * N + col] = v;
        }
    }
}

// ---------------------------------------------------------------------------
// Pack K (row-major [node][64]) and V transposed ([feat][node]) to bf16.
// ---------------------------------------------------------------------------
__global__ void k_pack_kv(const float* __restrict__ qkv, unsigned short* __restrict__ kb16,
                          unsigned short* __restrict__ vt16, int N)
{
    int i = blockIdx.x * blockDim.x + threadIdx.x;
    if (i >= N * 64) return;
    int node = i >> 6, f = i & 63;
    kb16[i]                    = f2bf(qkv[(size_t)node * 192 + 64 + f]);
    vt16[(size_t)f * N + node] = f2bf(qkv[(size_t)node * 192 + 128 + f]);
}

// ---------------------------------------------------------------------------
// Flash attention, single head, C = 64, dense N x N softmax.
// qkv (fp32) for Q; kb16 [N,64] bf16; vt16 [64,N] bf16. out: [N,64] fp32.
// Block: 64 threads (2 waves) -> 32 query rows per block, 16 per wave.
// Double-buffered K/V tiles staged with gfx1250 async global->LDS copies.
// ---------------------------------------------------------------------------
__global__ void k_attn(const float* __restrict__ qkv,
                       const unsigned short* __restrict__ kb16,
                       const unsigned short* __restrict__ vt16,
                       float* __restrict__ out, int N)
{
    __shared__ __align__(32) unsigned short ldsK[2][32 * 64];   // [key][feat]
    __shared__ __align__(32) unsigned short ldsVt[2][64 * 32];  // [feat][key]
    __shared__ __align__(32) unsigned short ldsP[2][16 * 32];   // per-wave P scratch

    const int tid  = threadIdx.x;
    const int wave = tid >> 5;
    const int lane = tid & 31;
    const int rowbase = blockIdx.x * 32 + wave * 16;
    const float qscale = 0.125f; // 1/sqrt(64)

    // Q fragments (resident): feats 0..31 and 32..63, pre-scaled
    Frag qa[2];
    {
        const int qrow = rowbase + (lane & 15);
        const int k0   = (lane >> 4) * 16;
        const float* qr = qkv + (size_t)qrow * 192;
        #pragma unroll
        for (int ch = 0; ch < 2; ++ch) {
            #pragma unroll
            for (int i = 0; i < 8; ++i) {
                qa[ch].s[i]     = f2bf(qr[ch * 32 + k0 + i] * qscale);
                qa[ch].s[i + 8] = f2bf(qr[ch * 32 + k0 + 8 + i] * qscale);
            }
        }
    }

    v8f oacc[4] = {};
    float mrow[8], lrow[8];
    #pragma unroll
    for (int r = 0; r < 8; ++r) { mrow[r] = -1e30f; lrow[r] = 0.f; }

    // async stage of one 32-key tile pair into buffer `buf`
    auto stage = [&](int buf, int kb) {
        // K tile: contiguous 2048 ushorts starting at kb*64
        #pragma unroll
        for (int part = 0; part < 4; ++part) {
            int off = (part * 64 + tid) * 8;              // ushort index, 16B granules
            async_copy_b128(&ldsK[buf][off], kb16 + (size_t)kb * 64 + off);
        }
        // Vt tile: per-feat rows of 32 keys
        #pragma unroll
        for (int part = 0; part < 4; ++part) {
            int idx = (part * 64 + tid) * 8;
            int f = idx >> 5, ko = idx & 31;
            async_copy_b128(&ldsVt[buf][f * 32 + ko], vt16 + (size_t)f * N + kb + ko);
        }
    };

    const int nTiles = N / 32;
    stage(0, 0);
    for (int kt = 0; kt < nTiles; ++kt) {
        const int buf = kt & 1;
        wait_async0();        // this wave's issued tile copies are in LDS
        __syncthreads();      // everyone's copies are in LDS; prev buf free
        if (kt + 1 < nTiles) stage(buf ^ 1, (kt + 1) * 32);

        // S = (Q*scale) @ K^T : two 16x16 tiles (keys 0-15, 16-31)
        v8f s0 = {}, s1 = {};
        {
            const int keyc = lane & 15;
            const int bk0  = (lane >> 4) * 16;
            Frag b;
            b.u = *(const v16u16*)&ldsK[buf][keyc * 64 + bk0];             s0 = wmma_bf16(qa[0], b, s0);
            b.u = *(const v16u16*)&ldsK[buf][keyc * 64 + 32 + bk0];        s0 = wmma_bf16(qa[1], b, s0);
            b.u = *(const v16u16*)&ldsK[buf][(16 + keyc) * 64 + bk0];      s1 = wmma_bf16(qa[0], b, s1);
            b.u = *(const v16u16*)&ldsK[buf][(16 + keyc) * 64 + 32 + bk0]; s1 = wmma_bf16(qa[1], b, s1);
        }

        // online softmax per row (lanes 0-15 own rows r, lanes 16-31 rows r+8)
        #pragma unroll
        for (int r = 0; r < 8; ++r) {
            float v0 = s0[r], v1 = s1[r];
            float mx = fmaxf(v0, v1);
            #pragma unroll
            for (int m = 1; m < 16; m <<= 1) mx = fmaxf(mx, __shfl_xor(mx, m, 32));
            float mnew = fmaxf(mrow[r], mx);
            float corr = __expf(mrow[r] - mnew);
            float p0 = __expf(v0 - mnew);
            float p1 = __expf(v1 - mnew);
            float rs = p0 + p1;
            #pragma unroll
            for (int m = 1; m < 16; m <<= 1) rs += __shfl_xor(rs, m, 32);
            lrow[r] = lrow[r] * corr + rs;
            mrow[r] = mnew;
            #pragma unroll
            for (int nt = 0; nt < 4; ++nt) oacc[nt][r] *= corr;
            int prow = r + ((lane >> 4) << 3);
            int pcol = lane & 15;
            ldsP[wave][prow * 32 + pcol]      = f2bf(p0);
            ldsP[wave][prow * 32 + pcol + 16] = f2bf(p1);
        }

        // P (16x32 over keys) as A fragment; accumulate O += P @ V
        Frag pa;
        {
            const int arow = lane & 15;
            const int ak0  = (lane >> 4) * 16;
            const unsigned short* pp = &ldsP[wave][arow * 32 + ak0];
            *(v8u16*)&pa.s[0] = *(const v8u16*)pp;
            *(v8u16*)&pa.s[8] = *(const v8u16*)(pp + 8);
        }
        #pragma unroll
        for (int nt = 0; nt < 4; ++nt) {
            Frag bv;
            const int f  = nt * 16 + (lane & 15);
            const int bk = (lane >> 4) * 16;
            bv.u = *(const v16u16*)&ldsVt[buf][f * 32 + bk];
            oacc[nt] = wmma_bf16(pa, bv, oacc[nt]);
        }
    }

    #pragma unroll
    for (int nt = 0; nt < 4; ++nt) {
        #pragma unroll
        for (int r = 0; r < 8; ++r) {
            int row  = rowbase + r + ((lane >> 4) << 3);
            int feat = nt * 16 + (lane & 15);
            out[(size_t)row * 64 + feat] = oacc[nt][r] / lrow[r];
        }
    }
}

// ---------------------------------------------------------------------------
// Scalar helper kernels
// ---------------------------------------------------------------------------
__global__ void k_deg_init(float* deg, int n) {
    int i = blockIdx.x * blockDim.x + threadIdx.x;
    if (i < n) deg[i] = 1.0f;  // self loop
}
__global__ void k_deg_edges(const int* __restrict__ dst, float* deg, int E) {
    int e = blockIdx.x * blockDim.x + threadIdx.x;
    if (e < E) atomicAdd(&deg[dst[e]], 1.0f);
}
__global__ void k_dinv(float* deg, int n) {
    int i = blockIdx.x * blockDim.x + threadIdx.x;
    if (i < n) deg[i] = rsqrtf(deg[i]);
}
__global__ void k_gcn_init(const float* __restrict__ xw, const float* __restrict__ dinv,
                           const float* __restrict__ b, float* __restrict__ out, int total) {
    int i = blockIdx.x * blockDim.x + threadIdx.x;
    if (i >= total) return;
    int node = i >> 6, c = i & 63;
    float di = dinv[node];
    out[i] = b[c] + xw[i] * di * di;
}
__global__ void k_gcn_scatter(const float* __restrict__ xw, const float* __restrict__ dinv,
                              const int* __restrict__ src, const int* __restrict__ dst,
                              float* __restrict__ out, int E) {
    long long idx = (long long)blockIdx.x * blockDim.x + threadIdx.x;
    if (idx >= (long long)E * 64) return;
    int e = (int)(idx >> 6), c = (int)(idx & 63);
    int s = src[e], d = dst[e];
    float coef = dinv[s] * dinv[d];
    atomicAdd(&out[(size_t)d * 64 + c], xw[(size_t)s * 64 + c] * coef);
}
__global__ void k_add(const float* __restrict__ a, const float* __restrict__ b,
                      float* __restrict__ c, int n) {
    int i = blockIdx.x * blockDim.x + threadIdx.x;
    if (i < n) c[i] = a[i] + b[i];
}
__global__ void k_colstats(const float* __restrict__ x, float* __restrict__ mean,
                           float* __restrict__ rstd, int rows, int C) {
    __shared__ float rs[256], rs2[256];
    int c = blockIdx.x, t = threadIdx.x;
    float s = 0.f, s2 = 0.f;
    for (int r = t; r < rows; r += 256) {
        float v = x[(size_t)r * C + c];
        s += v; s2 += v * v;
    }
    rs[t] = s; rs2[t] = s2;
    __syncthreads();
    for (int off = 128; off > 0; off >>= 1) {
        if (t < off) { rs[t] += rs[t + off]; rs2[t] += rs2[t + off]; }
        __syncthreads();
    }
    if (t == 0) {
        float m = rs[0] / rows;
        float var = rs2[0] / rows - m * m;
        mean[c] = m;
        rstd[c] = rsqrtf(var + 1e-5f);
    }
}
__global__ void k_bn_apply(const float* __restrict__ x, const float* __restrict__ mean,
                           const float* __restrict__ rstd, const float* __restrict__ g,
                           const float* __restrict__ b, float* __restrict__ y, int total) {
    int i = blockIdx.x * blockDim.x + threadIdx.x;
    if (i >= total) return;
    int c = i & 63;
    y[i] = (x[i] - mean[c]) * rstd[c] * g[c] + b[c];
}
__global__ void k_ln_relu(const float* __restrict__ x, const float* __restrict__ g,
                          const float* __restrict__ b, float* __restrict__ y, int N) {
    int gt = blockIdx.x * blockDim.x + threadIdx.x;
    int node = gt >> 5, lane = gt & 31;
    if (node >= N) return;
    const float* row = x + (size_t)node * 64;
    float v0 = row[lane], v1 = row[lane + 32];
    float s = v0 + v1, s2 = v0 * v0 + v1 * v1;
    #pragma unroll
    for (int m = 1; m < 32; m <<= 1) { s += __shfl_xor(s, m, 32); s2 += __shfl_xor(s2, m, 32); }
    float mean = s * (1.f / 64.f);
    float var  = s2 * (1.f / 64.f) - mean * mean;
    float r = rsqrtf(var + 1e-5f);
    float* yo = y + (size_t)node * 64;
    yo[lane]      = fmaxf((v0 - mean) * r * g[lane] + b[lane], 0.f);
    yo[lane + 32] = fmaxf((v1 - mean) * r * g[lane + 32] + b[lane + 32], 0.f);
}
__global__ void k_head(const float* __restrict__ h, const float* __restrict__ W,
                       const float* __restrict__ b, float* __restrict__ out, int N) {
    int n = blockIdx.x * blockDim.x + threadIdx.x;
    if (n >= N) return;
    const float* hr = h + (size_t)n * 64;
    float logits[21];
    #pragma unroll
    for (int j = 0; j < 21; ++j) logits[j] = b[j];
    for (int k = 0; k < 64; ++k) {
        float hv = hr[k];
        #pragma unroll
        for (int j = 0; j < 21; ++j) logits[j] += hv * W[k * 21 + j];
    }
    float mx = logits[0];
    #pragma unroll
    for (int j = 1; j < 21; ++j) mx = fmaxf(mx, logits[j]);
    float se = 0.f;
    #pragma unroll
    for (int j = 0; j < 21; ++j) se += __expf(logits[j] - mx);
    float lse = mx + logf(se);
    float* o = out + (size_t)n * 21;
    #pragma unroll
    for (int j = 0; j < 21; ++j) o[j] = logits[j] - lse;
}

// ---------------------------------------------------------------------------
// Orchestration
// ---------------------------------------------------------------------------
extern "C" void kernel_launch(void* const* d_in, const int* in_sizes, int n_in,
                              void* d_out, int out_size, void* d_ws, size_t ws_size,
                              hipStream_t stream)
{
    const int N = 8192, E = 131072, IN = 128, C = 64, L = 4;

    const float* x      = (const float*)d_in[0];
    const int*   ei     = (const int*)d_in[1];
    const float* W_in   = (const float*)d_in[2];
    const float* b_in   = (const float*)d_in[3];
    const float* gcn_W  = (const float*)d_in[4];
    const float* gcn_b  = (const float*)d_in[5];
    const float* Wqkv   = (const float*)d_in[6];
    const float* bqkv   = (const float*)d_in[7];
    const float* Wo     = (const float*)d_in[8];
    const float* bo     = (const float*)d_in[9];
    const float* bn1_g  = (const float*)d_in[10];
    const float* bn1_b  = (const float*)d_in[11];
    const float* bn2_g  = (const float*)d_in[12];
    const float* bn2_b  = (const float*)d_in[13];
    const float* bn3_g  = (const float*)d_in[14];
    const float* bn3_b  = (const float*)d_in[15];
    const float* mlp_W1 = (const float*)d_in[16];
    const float* mlp_b1 = (const float*)d_in[17];
    const float* mlp_W2 = (const float*)d_in[18];
    const float* mlp_b2 = (const float*)d_in[19];
    const float* ln_g   = (const float*)d_in[20];
    const float* ln_b   = (const float*)d_in[21];
    const float* W_out  = (const float*)d_in[22];
    const float* b_out  = (const float*)d_in[23];

    const int* src = ei;
    const int* dst = ei + E;

    // workspace carve-up
    float* p    = (float*)d_ws;
    float* dinv = p; p += N;
    float* h    = p; p += (size_t)N * C;
    float* t0   = p; p += (size_t)N * C;
    float* t1   = p; p += (size_t)N * C;
    float* t2   = p; p += (size_t)N * C;
    float* t3   = p; p += (size_t)N * C;
    float* qkv  = p; p += (size_t)N * 3 * C;
    float* hid  = p; p += (size_t)N * 2 * C;
    float* mean = p; p += C;
    float* rstd = p; p += C;
    unsigned short* kb16 = (unsigned short*)p; p += (size_t)N * C / 2;
    unsigned short* vt16 = (unsigned short*)p; p += (size_t)N * C / 2;

    const int NC = N * C;
    dim3 b256(256);

    // degrees -> dinv
    k_deg_init<<<N / 256, b256, 0, stream>>>(dinv, N);
    k_deg_edges<<<E / 256, b256, 0, stream>>>(dst, dinv, E);
    k_dinv<<<N / 256, b256, 0, stream>>>(dinv, N);

    // input transform: h = x @ W_in + b_in
    k_gemm<<<dim3(N / 64, C / 64), 128, 0, stream>>>(x, W_in, b_in, h, N, C, IN, 0, 0);

    for (int l = 0; l < L; ++l) {
        // ---- local GCN branch ----
        k_gemm<<<dim3(N / 64, 1), 128, 0, stream>>>(h, gcn_W + (size_t)l * C * C, nullptr,
                                                    t0, N, C, C, 0, 0);
        k_gcn_init<<<NC / 256, b256, 0, stream>>>(t0, dinv, gcn_b + (size_t)l * C, t1, NC);
        k_gcn_scatter<<<(int)(((long long)E * 64 + 255) / 256), b256, 0, stream>>>(
            t0, dinv, src, dst, t1, E);
        k_add<<<NC / 256, b256, 0, stream>>>(t1, h, t1, NC);
        k_colstats<<<C, b256, 0, stream>>>(t1, mean, rstd, N, C);
        k_bn_apply<<<NC / 256, b256, 0, stream>>>(t1, mean, rstd,
                                                  bn1_g + (size_t)l * C, bn1_b + (size_t)l * C,
                                                  t1, NC);
        // ---- global attention branch ----
        k_gemm<<<dim3(N / 64, 3), 128, 0, stream>>>(h, Wqkv + (size_t)l * 3 * C * C,
                                                    bqkv + (size_t)l * 3 * C, qkv,
                                                    N, 3 * C, C, /*bT=*/1, 0);
        k_pack_kv<<<NC / 256, b256, 0, stream>>>(qkv, kb16, vt16, N);
        k_attn<<<N / 32, 64, 0, stream>>>(qkv, kb16, vt16, t2, N);
        k_gemm<<<dim3(N / 64, 1), 128, 0, stream>>>(t2, Wo + (size_t)l * C * C,
                                                    bo + (size_t)l * C, t0,
                                                    N, C, C, /*bT=*/1, 0);
        k_add<<<NC / 256, b256, 0, stream>>>(t0, h, t2, NC);
        k_colstats<<<C, b256, 0, stream>>>(t2, mean, rstd, N, C);
        k_bn_apply<<<NC / 256, b256, 0, stream>>>(t2, mean, rstd,
                                                  bn2_g + (size_t)l * C, bn2_b + (size_t)l * C,
                                                  t2, NC);
        // ---- combine + MLP ----
        k_add<<<NC / 256, b256, 0, stream>>>(t1, t2, t3, NC);
        k_gemm<<<dim3(N / 64, 2), 128, 0, stream>>>(t3, mlp_W1 + (size_t)l * C * 2 * C,
                                                    mlp_b1 + (size_t)l * 2 * C, hid,
                                                    N, 2 * C, C, 0, /*relu=*/1);
        k_gemm<<<dim3(N / 64, 1), 128, 0, stream>>>(hid, mlp_W2 + (size_t)l * 2 * C * C,
                                                    mlp_b2 + (size_t)l * C, t0,
                                                    N, C, 2 * C, 0, 0);
        k_add<<<NC / 256, b256, 0, stream>>>(t3, t0, t1, NC);
        k_colstats<<<C, b256, 0, stream>>>(t1, mean, rstd, N, C);
        k_bn_apply<<<NC / 256, b256, 0, stream>>>(t1, mean, rstd,
                                                  bn3_g + (size_t)l * C, bn3_b + (size_t)l * C,
                                                  t1, NC);
        // ---- LayerNorm + ReLU -> h ----
        k_ln_relu<<<(N * 32) / 256, b256, 0, stream>>>(t1, ln_g + (size_t)l * C,
                                                       ln_b + (size_t)l * C, h, N);
    }

    // output head
    k_head<<<N / 256, b256, 0, stream>>>(h, W_out, b_out, (float*)d_out, N);

    (void)in_sizes; (void)n_in; (void)out_size; (void)ws_size;
}
